// MiddleLayerPathwayMLP_FiveLayer_33981781246569
// MI455X (gfx1250) — matrix operand order repaired and checked by reading, not compiled, not run on hardware
//
#include <hip/hip_runtime.h>
#include <cstdint>
#include <cstddef>

// ---------------- CDNA5 WMMA types ----------------
typedef __attribute__((ext_vector_type(16))) __bf16 v16bf;
typedef __attribute__((ext_vector_type(8)))  __bf16 v8bf;
typedef __attribute__((ext_vector_type(8)))  float  v8f;

__device__ __forceinline__ float gelu_exact(float x) {
  return 0.5f * x * (1.0f + erff(x * 0.70710678118654752f));
}

// CDNA5 async global->LDS (ASYNCcnt), 16 bytes per lane.
__device__ __forceinline__ void async_copy16(void* lds_dst, const void* gsrc) {
  asm volatile("global_load_async_to_lds_b128 %0, %1, off"
               :: "v"((unsigned)(uintptr_t)lds_dst), "v"(gsrc)
               : "memory");
}
__device__ __forceinline__ void wait_async0() {
  asm volatile("s_wait_asynccnt 0" ::: "memory");
}

// A fragment: 16x32 bf16 (MxK). Lane l: row = l&15; k-base = (l>>4)*8.
__device__ __forceinline__ v16bf load_frag_a(const __bf16* __restrict__ A, int lda,
                                             int row, int k0, int lane) {
  const int kb = (lane >> 4) << 3; // 0 or 8
  const __bf16* p = A + (size_t)row * lda + k0 + kb;
  v8bf lo = *(const v8bf*)p;        // k = kb .. kb+7
  v8bf hi = *(const v8bf*)(p + 16); // k = kb+16 .. kb+23
  v16bf r;
#pragma unroll
  for (int i = 0; i < 8; ++i) { r[i] = lo[i]; r[i + 8] = hi[i]; }
  return r;
}

// B fragment: 32x16 bf16 (KxN), from a row-major [16,K] panel (one row per out col).
// Lane l: col = col0 + (l&15); k-base = (l>>4)*8; k = kb..kb+7, kb+16..kb+23.
__device__ __forceinline__ v16bf load_frag_b(const __bf16* __restrict__ W, int ldw,
                                             int col0, int k0, int lane) {
  const int wrow = col0 + (lane & 15);
  const int kb = (lane >> 4) << 3; // 0 or 8
  const __bf16* p = W + (size_t)wrow * ldw + k0 + kb;
  v8bf lo = *(const v8bf*)p;
  v8bf hi = *(const v8bf*)(p + 16);
  v16bf r;
#pragma unroll
  for (int i = 0; i < 8; ++i) { r[i] = lo[i]; r[i + 8] = hi[i]; }
  return r;
}

// ---------------- convert + pad (f32 -> bf16) ----------------
__global__ __launch_bounds__(256) void convert_pad_kernel(
    const float* __restrict__ src, __bf16* __restrict__ dst,
    int srows, int scols, int dcols) {
  const int r = blockIdx.x;
  for (int c = threadIdx.x; c < dcols; c += blockDim.x) {
    float v = (r < srows && c < scols) ? src[(size_t)r * scols + c] : 0.0f;
    dst[(size_t)r * dcols + c] = (__bf16)v;
  }
}

// ---------------- generic GEMM: D = act(A[M,K] * W[N,K]^T + bias) ----------------
// 256 threads = 8 waves; each wave owns 16 rows; A panel held in VGPRs.
// B 16-col panels staged in LDS via async global->LDS, double-buffered,
// shared by all 8 waves (8x cut in L2 traffic vs per-wave streaming).
template <int K, bool GELU, bool STORE_BF16>
__global__ __launch_bounds__(256) void gemm_wmma(
    const __bf16* __restrict__ A, int lda,
    const __bf16* __restrict__ W, int ldw,
    const float* __restrict__ bias,
    void* __restrict__ Cout, int ldc,
    int ntiles, int nvalid) {
  constexpr int KT = K / 32;
  constexpr int KCH = K / 8;        // 16B chunks per panel row
  constexpr int CHUNKS = 16 * KCH;  // 16B chunks per 16-col panel
  __shared__ __bf16 Bs[2][16 * K];

  const int lane = threadIdx.x & 31;
  const int wave = threadIdx.x >> 5;
  const int row0 = (blockIdx.x * 8 + wave) * 16;
  const int arow = row0 + (lane & 15);

  // Load the whole 16xK A panel once (reused across the full N sweep).
  v16bf a[KT];
#pragma unroll
  for (int k = 0; k < KT; ++k) a[k] = load_frag_a(A, lda, arow, k * 32, lane);

  auto issue_panel = [&](int t, int buf) {
    for (int i = threadIdx.x; i < CHUNKS; i += 256) {
      const int colr = i / KCH;
      const int kk = (i - colr * KCH) * 8;
      const __bf16* g = W + (size_t)(t * 16 + colr) * ldw + kk;
      async_copy16(&Bs[buf][colr * K + kk], g);
    }
  };

  issue_panel(0, 0);
  wait_async0();
  __syncthreads();

  const int col = lane & 15;
  const int rbase = (lane >> 4) << 3;

  for (int t = 0; t < ntiles; ++t) {
    const int buf = t & 1;
    if (t + 1 < ntiles) issue_panel(t + 1, buf ^ 1); // prefetch next panel async

    v8f acc = {};
#pragma unroll
    for (int k = 0; k < KT; ++k) {
      v16bf b = load_frag_b(&Bs[buf][0], K, 0, k * 32, lane); // ds_load_b128 x2
      acc = __builtin_amdgcn_wmma_f32_16x16x32_bf16(
          false, a[k], false, b, (short)0, acc, false, false);
    }

    const int c = t * 16 + col;
    if (c < nvalid) {
      const float bv = bias[c];
#pragma unroll
      for (int r = 0; r < 8; ++r) {
        float v = acc[r] + bv;
        if (GELU) v = gelu_exact(v);
        const int row = row0 + rbase + r;
        if (STORE_BF16)
          ((__bf16*)Cout)[(size_t)row * ldc + c] = (__bf16)v;
        else
          ((float*)Cout)[(size_t)row * ldc + c] = v;
      }
    }

    wait_async0();     // this wave's slice of next panel landed
    __syncthreads();   // whole panel visible to all 8 waves
  }
}

// ---------------- fused routed middle layer ----------------
// mid_out[b, o*32+n] = gelu( sum_g pw[b,g,o] * (xi_g . W3[o*32+n, g*64:+64]) + (sum_g pw)*b3 )
// pw = softmax(logits[b, 0:16]) viewed as [GI=4, GO=4].
// Implemented by scaling A fragments by pw[g,o] pre-WMMA; 4 waves/block, 16 rows/wave.
__global__ __launch_bounds__(128) void middle_kernel(
    const __bf16* __restrict__ X,    // [M,256] bf16 (mid_in)
    const __bf16* __restrict__ W3b,  // [128,256] bf16
    const float* __restrict__ b3,    // [128]
    const float* __restrict__ logits,// [M,16] f32
    __bf16* __restrict__ out) {      // [M,128] bf16
  const int lane = threadIdx.x & 31;
  const int wave = threadIdx.x >> 5;
  const int row0 = (blockIdx.x * 4 + wave) * 16;
  const int myrow = row0 + (lane & 15);

  // per-lane softmax of this lane's row (both lane halves hold the same row)
  float lg[16];
  const float* lp = logits + (size_t)myrow * 16;
  float mx = -3.0e38f;
#pragma unroll
  for (int i = 0; i < 16; ++i) { lg[i] = lp[i]; mx = fmaxf(mx, lg[i]); }
  float sum = 0.0f;
#pragma unroll
  for (int i = 0; i < 16; ++i) { lg[i] = expf(lg[i] - mx); sum += lg[i]; }
  const float inv = 1.0f / sum;
#pragma unroll
  for (int i = 0; i < 16; ++i) lg[i] *= inv; // pw[g*4+o]
  float pwsum[4];
#pragma unroll
  for (int o = 0; o < 4; ++o) pwsum[o] = lg[o] + lg[4 + o] + lg[8 + o] + lg[12 + o];

  v8f acc[4][2];
#pragma unroll
  for (int o = 0; o < 4; ++o)
#pragma unroll
    for (int nt = 0; nt < 2; ++nt) { v8f z = {}; acc[o][nt] = z; }

#pragma unroll
  for (int g = 0; g < 4; ++g) {
    v16bf a0 = load_frag_a(X, 256, myrow, g * 64, lane);
    v16bf a1 = load_frag_a(X, 256, myrow, g * 64 + 32, lane);
#pragma unroll
    for (int o = 0; o < 4; ++o) {
      const float s = lg[g * 4 + o];
      v16bf s0, s1;
#pragma unroll
      for (int i = 0; i < 16; ++i) {
        s0[i] = (__bf16)((float)a0[i] * s);
        s1[i] = (__bf16)((float)a1[i] * s);
      }
#pragma unroll
      for (int nt = 0; nt < 2; ++nt) {
        const int col0 = o * 32 + nt * 16;
        v16bf b0 = load_frag_b(W3b, 256, col0, g * 64, lane);
        v16bf b1 = load_frag_b(W3b, 256, col0, g * 64 + 32, lane);
        acc[o][nt] = __builtin_amdgcn_wmma_f32_16x16x32_bf16(
            false, s0, false, b0, (short)0, acc[o][nt], false, false);
        acc[o][nt] = __builtin_amdgcn_wmma_f32_16x16x32_bf16(
            false, s1, false, b1, (short)0, acc[o][nt], false, false);
      }
    }
  }

  const int col = lane & 15;
  const int rbase = (lane >> 4) << 3;
#pragma unroll
  for (int o = 0; o < 4; ++o) {
#pragma unroll
    for (int nt = 0; nt < 2; ++nt) {
      const int c = o * 32 + nt * 16 + col;
      const float bv = b3[c];
#pragma unroll
      for (int r = 0; r < 8; ++r) {
        const int rl = rbase + r; // local row index 0..15 (held by lane rl)
        const float ps = __shfl(pwsum[o], rl, 32);
        float v = acc[o][nt][r] + ps * bv;
        v = gelu_exact(v);
        out[(size_t)(row0 + rl) * 128 + c] = (__bf16)v;
      }
    }
  }
}

// ---------------- host side ----------------
static inline size_t align256(size_t x) { return (x + 255) & ~(size_t)255; }

extern "C" void kernel_launch(void* const* d_in, const int* in_sizes, int n_in,
                              void* d_out, int out_size, void* d_ws, size_t ws_size,
                              hipStream_t stream) {
  (void)n_in; (void)out_size; (void)ws_size;

  const float* x  = (const float*)d_in[0];
  const float* W1 = (const float*)d_in[1];
  const float* b1 = (const float*)d_in[2];
  const float* W2 = (const float*)d_in[3];
  const float* b2 = (const float*)d_in[4];
  const float* W3 = (const float*)d_in[5];
  const float* b3 = (const float*)d_in[6];
  const float* W4 = (const float*)d_in[7];
  const float* b4 = (const float*)d_in[8];
  const float* W5 = (const float*)d_in[9];
  const float* b5 = (const float*)d_in[10];
  const float* W6 = (const float*)d_in[11];
  const float* b6 = (const float*)d_in[12];
  const float* Wr = (const float*)d_in[13];
  const float* br = (const float*)d_in[14];

  const int M = in_sizes[0] / 784; // 131072

  char* ws = (char*)d_ws;
  size_t off = 0;
  auto alloc = [&](size_t bytes) { size_t o = off; off = align256(off + bytes); return o; };

  __bf16* xb   = (__bf16*)(ws + alloc((size_t)M * 800 * 2));
  __bf16* h1b  = (__bf16*)(ws + alloc((size_t)M * 512 * 2));
  __bf16* h2b  = (__bf16*)(ws + alloc((size_t)M * 256 * 2));
  __bf16* midb = (__bf16*)(ws + alloc((size_t)M * 128 * 2));
  __bf16* h4b  = (__bf16*)(ws + alloc((size_t)M * 64 * 2));
  __bf16* h5b  = (__bf16*)(ws + alloc((size_t)M * 32 * 2));
  float*  lg   = (float*)(ws + alloc((size_t)M * 16 * 4));
  __bf16* W1b  = (__bf16*)(ws + alloc((size_t)512 * 800 * 2));
  __bf16* W2b  = (__bf16*)(ws + alloc((size_t)256 * 512 * 2));
  __bf16* W3b  = (__bf16*)(ws + alloc((size_t)128 * 256 * 2));
  __bf16* W4b  = (__bf16*)(ws + alloc((size_t)64 * 128 * 2));
  __bf16* W5b  = (__bf16*)(ws + alloc((size_t)32 * 64 * 2));
  __bf16* W6b  = (__bf16*)(ws + alloc((size_t)16 * 32 * 2));
  __bf16* Wrb  = (__bf16*)(ws + alloc((size_t)16 * 256 * 2));

  // conversions (bf16, K padded to multiples of 32 where needed)
  convert_pad_kernel<<<M, 256, 0, stream>>>(x, xb, M, 784, 800);
  convert_pad_kernel<<<512, 256, 0, stream>>>(W1, W1b, 512, 784, 800);
  convert_pad_kernel<<<256, 256, 0, stream>>>(W2, W2b, 256, 512, 512);
  convert_pad_kernel<<<128, 256, 0, stream>>>(W3, W3b, 128, 256, 256);
  convert_pad_kernel<<<64, 256, 0, stream>>>(W4, W4b, 64, 128, 128);
  convert_pad_kernel<<<32, 256, 0, stream>>>(W5, W5b, 32, 64, 64);
  convert_pad_kernel<<<16, 256, 0, stream>>>(W6, W6b, 10, 32, 32);   // row-pad 10->16
  convert_pad_kernel<<<16, 256, 0, stream>>>(Wr, Wrb, 16, 256, 256);

  const int gblocks = M / 128; // 8 waves * 16 rows per block
  // L1: [M,800]x[512,800]^T -> gelu -> bf16 [M,512]
  gemm_wmma<800, true, true><<<gblocks, 256, 0, stream>>>(xb, 800, W1b, 800, b1, h1b, 512, 32, 512);
  // L2: -> gelu -> bf16 [M,256]
  gemm_wmma<512, true, true><<<gblocks, 256, 0, stream>>>(h1b, 512, W2b, 512, b2, h2b, 256, 16, 256);
  // router logits: f32 [M,16]
  gemm_wmma<256, false, false><<<gblocks, 256, 0, stream>>>(h2b, 256, Wrb, 256, br, lg, 16, 1, 16);
  // fused routed middle layer -> bf16 [M,128]
  middle_kernel<<<M / 64, 128, 0, stream>>>(h2b, W3b, b3, lg, midb);
  // L4, L5
  gemm_wmma<128, true, true><<<gblocks, 256, 0, stream>>>(midb, 128, W4b, 128, b4, h4b, 64, 4, 64);
  gemm_wmma<64, true, true><<<gblocks, 256, 0, stream>>>(h4b, 64, W5b, 64, b5, h5b, 32, 2, 32);
  // L6: f32 out [M,10] (N padded to 16, store-masked to 10)
  gemm_wmma<32, false, false><<<gblocks, 256, 0, stream>>>(h5b, 32, W6b, 32, b6, d_out, 10, 1, 10);
}
